// AdaptiveRankLSTM_30365418783339
// MI455X (gfx1250) — matrix operand order repaired and checked by reading, not compiled, not run on hardware
//
#include <hip/hip_runtime.h>
#include <stdint.h>
#include <stddef.h>

// ---------------------------------------------------------------------------
// TT-LSTM for gfx1250 (MI455X).  TT cores collapse to a rank-16 bottleneck:
//   tt_matvec(x) = (x @ A[1024x16]) @ B[16x4096]
// Phase 1: build A_ih, A_hh, Bcat=[B_ih;B_hh] (bf16, pre-swizzled into WMMA
//          per-lane fragment order) and bias = b_ih + b_hh.
// Phase 2: ih_r = X @ A_ih   (32768x1024 * 1024x16, bf16 WMMA, f32 acc)
// Phase 3: persistent single-workgroup recurrence, 32 waves:
//          hh_r = h @ A_hh (WMMA + LDS f32 atomics reduce)
//          gates = [ih_r_t | hh_r] @ Bcat + bias  (WMMA)
//          LSTM pointwise in f32; h -> LDS bf16 + global out; c in VGPRs.
// All f32->bf16 conversions use the native __bf16 cast (v_cvt_pk_bf16_f32)
// instead of software RNE bit-twiddling.
// ---------------------------------------------------------------------------

typedef __attribute__((ext_vector_type(16))) __bf16 v16bf;
typedef __attribute__((ext_vector_type(8)))  float  v8f;

#define T_SEQ 512
#define BATCH 64
#define DH    1024

// workspace byte offsets
#define WS_AIH  0                    // 1024x16 bf16  (packed B-frag order) 32KB
#define WS_AHH  (32*1024)            // 32KB
#define WS_PB   (64*1024)            // 32x4096 bf16 packed (256 tiles)    256KB
#define WS_BIAS (320*1024)           // 4096 f32                            16KB
#define WS_IHR  (336*1024)           // 32768x16 bf16                        1MB

struct __align__(16) U4 { unsigned int x, y, z, w; };

union FragBF {
    v16bf v;
    U4 q[2];
    __bf16 h[16];
};

__device__ inline __bf16 to_bf16(float f) { return (__bf16)f; }   // hw cvt
__device__ inline float sigmoid_f(float x) { return 1.0f / (1.0f + __expf(-x)); }
__device__ inline float tanh_f(float x)    { return 1.0f - 2.0f / (__expf(2.0f * x) + 1.0f); }

// ---- Phase 1a: A[k,d] = sum_{a,c} g1[i,a] g2[a,j,c] g3[c,kk,d]; k=(i,j,kk)
// stored in WMMA B-matrix fragment order: lane = 16*((k%32/8)%2) + d, elem = k%8 + 8*(k%32/16)
__global__ __launch_bounds__(256) void k_build_A(
    const float* __restrict__ g1, const float* __restrict__ g2,
    const float* __restrict__ g3, unsigned short* __restrict__ packedA_) {
    __bf16* packedA = (__bf16*)packedA_;
    int idx = blockIdx.x * 256 + threadIdx.x;       // 0..16383
    int k = idx >> 4, d = idx & 15;
    int i = k >> 7, jj = (k >> 4) & 7, kk = k & 15;
    float acc = 0.f;
    for (int a = 0; a < 16; ++a) {
        float inner = 0.f;
        for (int c = 0; c < 16; ++c)
            inner += g2[(a * 8 + jj) * 16 + c] * g3[(c * 16 + kk) * 16 + d];
        acc += g1[i * 16 + a] * inner;
    }
    int q = k >> 5, rl = k & 31, hi = (rl >> 3) & 1, id = (rl & 7) | ((rl >> 4) << 3);
    packedA[(q * 32 + hi * 16 + d) * 16 + id] = to_bf16(acc);
}

// ---- Phase 1b: Bcat[dg,col]; rows 0-15 from ih cores, 16-31 from hh cores.
// col=(n,m,p).  Packed per 16-col tile in B-frag order.
__global__ __launch_bounds__(256) void k_build_B(
    const float* __restrict__ g4i, const float* __restrict__ g5i, const float* __restrict__ g6i,
    const float* __restrict__ g4h, const float* __restrict__ g5h, const float* __restrict__ g6h,
    unsigned short* __restrict__ packedB_) {
    __bf16* packedB = (__bf16*)packedB_;
    int idx = blockIdx.x * 256 + threadIdx.x;       // 0..131071
    int dg  = idx >> 12;                            // 0..31
    int col = idx & 4095;
    const float *g4, *g5, *g6; int d;
    if (dg < 16) { g4 = g4i; g5 = g5i; g6 = g6i; d = dg; }
    else         { g4 = g4h; g5 = g5h; g6 = g6h; d = dg - 16; }
    int n = col >> 8, m = (col >> 4) & 15, p = col & 15;
    float acc = 0.f;
    for (int e = 0; e < 16; ++e) {
        float inner = 0.f;
        for (int f = 0; f < 16; ++f)
            inner += g5[(e * 16 + m) * 16 + f] * g6[f * 16 + p];
        acc += g4[(d * 16 + n) * 16 + e] * inner;
    }
    int tile = col >> 4, hi = (dg >> 3) & 1, id = (dg & 7) | ((dg >> 4) << 3);
    packedB[(tile * 32 + hi * 16 + (col & 15)) * 16 + id] = to_bf16(acc);
}

__global__ __launch_bounds__(256) void k_bias(
    const float* __restrict__ bi, const float* __restrict__ bh, float* __restrict__ bias) {
    int i = blockIdx.x * 256 + threadIdx.x;
    bias[i] = bi[i] + bh[i];
}

// ---- Phase 2: ih_r = X @ A_ih.  One wave per 16-row tile; K=1024 in 32
// chunks of 16x16x32 bf16 WMMA with f32 accumulation.
__global__ __launch_bounds__(256) void k_ihr(
    const float* __restrict__ x, const unsigned short* __restrict__ pAih,
    unsigned short* __restrict__ ihr_) {
    __bf16* ihr = (__bf16*)ihr_;
    int gw   = (blockIdx.x * 256 + threadIdx.x) >> 5;   // tile 0..2047
    int lane = threadIdx.x & 31;
    int n = lane & 15, hi = lane >> 4;
    const float* xrow = x + (size_t)(gw * 16 + n) * 1024;
    const U4* pb = (const U4*)pAih;
    v8f acc = {0.f, 0.f, 0.f, 0.f, 0.f, 0.f, 0.f, 0.f};
    for (int q = 0; q < 32; ++q) {
        FragBF a, b;
        // A layout: elem[8s+j] = x[row, q*32 + 16s + 8*hi + j]
        const float4* p0 = (const float4*)(xrow + q * 32 + 8 * hi);
        const float4* p1 = (const float4*)(xrow + q * 32 + 16 + 8 * hi);
        float4 u0 = p0[0], u1 = p0[1], u2 = p1[0], u3 = p1[1];
        a.h[0]  = to_bf16(u0.x); a.h[1]  = to_bf16(u0.y);
        a.h[2]  = to_bf16(u0.z); a.h[3]  = to_bf16(u0.w);
        a.h[4]  = to_bf16(u1.x); a.h[5]  = to_bf16(u1.y);
        a.h[6]  = to_bf16(u1.z); a.h[7]  = to_bf16(u1.w);
        a.h[8]  = to_bf16(u2.x); a.h[9]  = to_bf16(u2.y);
        a.h[10] = to_bf16(u2.z); a.h[11] = to_bf16(u2.w);
        a.h[12] = to_bf16(u3.x); a.h[13] = to_bf16(u3.y);
        a.h[14] = to_bf16(u3.z); a.h[15] = to_bf16(u3.w);
        b.q[0] = pb[(q * 32 + lane) * 2 + 0];
        b.q[1] = pb[(q * 32 + lane) * 2 + 1];
        acc = __builtin_amdgcn_wmma_f32_16x16x32_bf16(false, a.v, false, b.v,
                                                      (short)0, acc, false, false);
    }
#pragma unroll
    for (int v = 0; v < 8; ++v)
        ihr[(size_t)(gw * 16 + v + 8 * hi) * 16 + n] = to_bf16(acc[v]);
}

// ---- Phase 3: persistent recurrence, 1 workgroup, 32 waves (wave32).
// wave w: r = w>>3 row-tile (16 batch rows), jw = w&7 -> cols 128*jw..+127.
__global__ __launch_bounds__(1024) void k_recur(
    const unsigned short* __restrict__ ihr_,  // (B*T) x 16 bf16
    const unsigned short* __restrict__ pAhh,  // packed A_hh fragments
    const unsigned short* __restrict__ pB,    // packed Bcat fragments
    const float* __restrict__ bias,           // 4096
    float* __restrict__ out) {
    const __bf16* ihr = (const __bf16*)ihr_;
    extern __shared__ unsigned char smem[];
    __bf16* h_bf   = (__bf16*)smem;                      // 64x1024 bf16 128KB
    __bf16* hr_cat = (__bf16*)(smem + 131072);           // 64x32 bf16    4KB
    float*  hr_acc = (float*)(smem + 131072 + 4096);     // 64x16 f32     4KB

    const int tid  = threadIdx.x;
    const int lane = tid & 31, w = tid >> 5;
    const int n = lane & 15, hi = lane >> 4;
    const int r = w >> 3, jw = w & 7;

    for (int i = tid; i < BATCH * DH; i += 1024) h_bf[i] = (__bf16)0.0f;
    hr_acc[tid & 1023] = 0.f;
    __syncthreads();

    v8f c_reg[8];
#pragma unroll
    for (int cc = 0; cc < 8; ++cc)
        c_reg[cc] = (v8f){0.f, 0.f, 0.f, 0.f, 0.f, 0.f, 0.f, 0.f};

    const U4* pb4 = (const U4*)pB;
    const U4* pa4 = (const U4*)pAhh;

    for (int t = 0; t < T_SEQ; ++t) {
        // prefetch next step's ih_r rows into cache (global_prefetch path)
        if (t + 1 < T_SEQ && tid < BATCH)
            __builtin_prefetch(&ihr[((size_t)tid * T_SEQ + t + 1) * 16], 0, 1);

        // ---- step 1: partial hh_r = h[16r:16r+16, 128jw:+128] @ A_hh[...]
        {
            v8f p = {0.f, 0.f, 0.f, 0.f, 0.f, 0.f, 0.f, 0.f};
#pragma unroll
            for (int q = 0; q < 4; ++q) {
                FragBF a, b;
                int kbase = 128 * jw + 32 * q;
                const U4* hp = (const U4*)(h_bf + (16 * r + n) * DH + kbase + 8 * hi);
                a.q[0] = hp[0];          // elems k = kbase + 8hi + 0..7
                a.q[1] = hp[2];          // elems k = kbase + 16 + 8hi + 0..7
                int qg = 4 * jw + q;
                b.q[0] = pa4[(qg * 32 + lane) * 2 + 0];
                b.q[1] = pa4[(qg * 32 + lane) * 2 + 1];
                p = __builtin_amdgcn_wmma_f32_16x16x32_bf16(false, a.v, false, b.v,
                                                            (short)0, p, false, false);
            }
#pragma unroll
            for (int v = 0; v < 8; ++v)
                atomicAdd(&hr_acc[(16 * r + v + 8 * hi) * 16 + n], p[v]);
        }
        __syncthreads();

        // ---- build hr_cat = [ih_r_t | hh_r]  (64 x 32 bf16)
        for (int e = tid; e < BATCH * 32; e += 1024) {
            int row = e >> 5, col = e & 31;
            if (col < 16)
                hr_cat[e] = ihr[((size_t)row * T_SEQ + t) * 16 + col];
            else
                hr_cat[e] = to_bf16(hr_acc[row * 16 + (col - 16)]);
        }
        __syncthreads();
        hr_acc[tid & 1023] = 0.f;   // re-zero for next step (no reader until next step1)

        // ---- step 2+3: gates = hr_cat @ Bcat + bias ; LSTM pointwise
        FragBF ha;
        {
            const U4* hp = (const U4*)(hr_cat + (16 * r + n) * 32 + 8 * hi);
            ha.q[0] = hp[0];
            ha.q[1] = hp[2];
        }
        for (int cc = 0; cc < 8; ++cc) {
            int ct = 8 * jw + cc;               // global 16-col tile, 0..63
            v8f acc_g[4];
#pragma unroll
            for (int g = 0; g < 4; ++g) {
                float bv = bias[g * 1024 + ct * 16 + n];
                v8f acc;
#pragma unroll
                for (int e = 0; e < 8; ++e) acc[e] = bv;
                FragBF b;
                int tile = g * 64 + ct;
                b.q[0] = pb4[(tile * 32 + lane) * 2 + 0];
                b.q[1] = pb4[(tile * 32 + lane) * 2 + 1];
                acc_g[g] = __builtin_amdgcn_wmma_f32_16x16x32_bf16(false, ha.v, false, b.v,
                                                                   (short)0, acc, false, false);
            }
            v8f cold = c_reg[cc], cn, hn;
#pragma unroll
            for (int e = 0; e < 8; ++e) {
                float iv = sigmoid_f(acc_g[0][e]);
                float fv = sigmoid_f(acc_g[1][e]);
                float gv = tanh_f(acc_g[2][e]);
                float ov = sigmoid_f(acc_g[3][e]);
                float cv = fv * cold[e] + iv * gv;
                cn[e] = cv;
                hn[e] = ov * tanh_f(cv);
            }
            c_reg[cc] = cn;
#pragma unroll
            for (int e = 0; e < 8; ++e) {
                int brow = 16 * r + e + 8 * hi;
                int col  = ct * 16 + n;
                float hv = hn[e];
                h_bf[brow * DH + col] = to_bf16(hv);
                out[(size_t)brow * (T_SEQ * DH) + (size_t)t * DH + col] = hv;
                if (t == T_SEQ - 1) {
                    size_t base = (size_t)BATCH * T_SEQ * DH;
                    out[base + (size_t)brow * DH + col] = hv;
                    out[base + (size_t)BATCH * DH + (size_t)brow * DH + col] = cn[e];
                }
            }
        }
        __syncthreads();   // h_bf / hr_acc ready for next step
    }
}

extern "C" void kernel_launch(void* const* d_in, const int* in_sizes, int n_in,
                              void* d_out, int out_size, void* d_ws, size_t ws_size,
                              hipStream_t stream) {
    (void)in_sizes; (void)n_in; (void)out_size; (void)ws_size;
    const float* x    = (const float*)d_in[0];
    const float* ig1  = (const float*)d_in[1];
    const float* ig2  = (const float*)d_in[2];
    const float* ig3  = (const float*)d_in[3];
    const float* ig4  = (const float*)d_in[4];
    const float* ig5  = (const float*)d_in[5];
    const float* ig6  = (const float*)d_in[6];
    const float* hg1  = (const float*)d_in[7];
    const float* hg2  = (const float*)d_in[8];
    const float* hg3  = (const float*)d_in[9];
    const float* hg4  = (const float*)d_in[10];
    const float* hg5  = (const float*)d_in[11];
    const float* hg6  = (const float*)d_in[12];
    const float* b_ih = (const float*)d_in[13];
    const float* b_hh = (const float*)d_in[14];

    unsigned char* ws = (unsigned char*)d_ws;
    unsigned short* pAih = (unsigned short*)(ws + WS_AIH);
    unsigned short* pAhh = (unsigned short*)(ws + WS_AHH);
    unsigned short* pB   = (unsigned short*)(ws + WS_PB);
    float*          bias = (float*)(ws + WS_BIAS);
    unsigned short* ihr  = (unsigned short*)(ws + WS_IHR);

    k_build_A<<<64, 256, 0, stream>>>(ig1, ig2, ig3, pAih);
    k_build_A<<<64, 256, 0, stream>>>(hg1, hg2, hg3, pAhh);
    k_build_B<<<512, 256, 0, stream>>>(ig4, ig5, ig6, hg4, hg5, hg6, pB);
    k_bias<<<16, 256, 0, stream>>>(b_ih, b_hh, bias);
    k_ihr<<<256, 256, 0, stream>>>(x, pAih, ihr);

    size_t smem = 131072 + 4096 + 4096;   // 136KB dynamic LDS (WGP has 320KB)
    (void)hipFuncSetAttribute((const void*)k_recur,
                              hipFuncAttributeMaxDynamicSharedMemorySize, (int)smem);
    k_recur<<<1, 1024, smem, stream>>>(ihr, pAhh, pB, bias, (float*)d_out);
}
